// Matcher_33406255629017
// MI455X (gfx1250) — matrix operand order repaired and checked by reading, not compiled, not run on hardware
//
#include <hip/hip_runtime.h>

#define HEADS 8
#define HEAD_DIM 64
#define CIN 256
#define N_TOK 4096     // 64*64
#define CMID 512       // HEADS*HEAD_DIM
#define OQKV 1536      // 3*CMID

typedef __attribute__((ext_vector_type(16))) __bf16 v16bf;
typedef __attribute__((ext_vector_type(8)))  float  v8f;

union FragU { v16bf v; uint4 q[2]; };
union PackU { uint4 q; unsigned short s[8]; };

__device__ __forceinline__ unsigned short f32_to_bf16(float f) {
  union { float f; unsigned u; } v; v.f = f;
  unsigned r = v.u + 0x7FFFu + ((v.u >> 16) & 1u);
  return (unsigned short)(r >> 16);
}

__device__ __forceinline__ v8f wmma_bf16(v16bf a, v16bf b, v8f c) {
  return __builtin_amdgcn_wmma_f32_16x16x32_bf16(false, a, false, b, (short)0, c, false, false);
}

// A fragment (16x32): source row-major, rows = M. Lane l<16: row l, K {0..7}U{16..23};
// lane l>=16: row l-16, K {8..15}U{24..31}. Two contiguous 16B loads per lane.
__device__ __forceinline__ v16bf load_frag_A(const unsigned short* Mp, int ld, int row0, int col0, int lane) {
  const int half = lane >> 4, m = lane & 15;
  const unsigned short* p = Mp + (size_t)(row0 + m) * ld + col0 + half * 8;
  FragU f;
  f.q[0] = *(const uint4*)(p);
  f.q[1] = *(const uint4*)(p + 16);
  return f.v;
}

// B fragment (32x16) where memory holds Mem[N][K] row-major (rows indexed by N).
// Lane l<16: col N=l, K {0..7}U{16..23}; lane l>=16: col N=l-16, K {8..15}U{24..31}. Two contiguous 16B loads per lane.
__device__ __forceinline__ v16bf load_frag_T(const unsigned short* Mp, int ld, int rowN0, int colK0, int lane) {
  const int half = lane >> 4, n = lane & 15;
  const unsigned short* p = Mp + (size_t)(rowN0 + n) * ld + colK0 + half * 8;
  FragU f;
  f.q[0] = *(const uint4*)(p);
  f.q[1] = *(const uint4*)(p + 16);
  return f.v;
}

// ---- CDNA5 async memory -> LDS (ASYNCcnt-tracked) ----
__device__ __forceinline__ void async_load_b128(unsigned lds_off, const void* gaddr) {
  asm volatile("global_load_async_to_lds_b128 %0, %1, off" :: "v"(lds_off), "v"(gaddr) : "memory");
}
__device__ __forceinline__ void wait_async0() {
  asm volatile("s_wait_asynccnt 0" ::: "memory");
}

// ---------------- conversion kernels ----------------

__global__ void k_transpose_x_bf16(const float* __restrict__ x, unsigned short* __restrict__ Xt) {
  int i = blockIdx.x * blockDim.x + threadIdx.x;   // over 256*4096
  int c = i >> 12, n = i & 4095;
  Xt[n * CIN + c] = f32_to_bf16(x[i]);
}

__global__ void k_cvt_bf16(const float* __restrict__ s, unsigned short* __restrict__ d, int nElem) {
  int i = blockIdx.x * blockDim.x + threadIdx.x;
  if (i < nElem) d[i] = f32_to_bf16(s[i]);
}

// ---------------- QKV projection (32x32 per wave) ----------------
// QKV[o][n] = sum_c Wq[o][c] * X[c][n]; o<1024 (Q,K) stored transposed into QKt[n][o],
// o>=1024 (V) stored natural into Vbuf[o-1024][n].
__global__ void __launch_bounds__(128) k_gemm_qkv(
    const unsigned short* __restrict__ Wq,   // 1536x256
    const unsigned short* __restrict__ Xt,   // 4096x256 (= X transposed)
    unsigned short* __restrict__ QKt,        // 4096x1024
    unsigned short* __restrict__ Vbuf) {     // 512x4096
  const int lane = threadIdx.x & 31;
  const int wave = threadIdx.x >> 5;
  const int tile = blockIdx.x * 4 + wave;    // 48 x 128 tiles of 32x32
  const int oBase = (tile >> 7) * 32;
  const int n0 = (tile & 127) * 32;
  v8f acc[2][2];
  v8f vz = {};
  acc[0][0] = vz; acc[0][1] = vz; acc[1][0] = vz; acc[1][1] = vz;
  #pragma unroll
  for (int kb = 0; kb < CIN; kb += 32) {
    v16bf a0 = load_frag_A(Wq, CIN, oBase,      kb, lane);
    v16bf a1 = load_frag_A(Wq, CIN, oBase + 16, kb, lane);
    v16bf b0 = load_frag_T(Xt, CIN, n0,      kb, lane);
    v16bf b1 = load_frag_T(Xt, CIN, n0 + 16, kb, lane);
    acc[0][0] = wmma_bf16(a0, b0, acc[0][0]);
    acc[0][1] = wmma_bf16(a0, b1, acc[0][1]);
    acc[1][0] = wmma_bf16(a1, b0, acc[1][0]);
    acc[1][1] = wmma_bf16(a1, b1, acc[1][1]);
  }
  const int half = lane >> 4;
  #pragma unroll
  for (int i = 0; i < 2; ++i) {
    const int oB = oBase + i * 16;
    #pragma unroll
    for (int j = 0; j < 2; ++j) {
      const int n = n0 + j * 16 + (lane & 15);
      if (oB < 1024) {
        PackU pk;
        #pragma unroll
        for (int r = 0; r < 8; ++r) pk.s[r] = f32_to_bf16(acc[i][j][r]);
        *(uint4*)(QKt + (size_t)n * 1024 + oB + half * 8) = pk.q;
      } else {
        const int vrow = oB - 1024 + half * 8;
        #pragma unroll
        for (int r = 0; r < 8; ++r)
          Vbuf[(size_t)(vrow + r) * N_TOK + n] = f32_to_bf16(acc[i][j][r]);
      }
    }
  }
}

// ---------------- flash attention ----------------
// Workgroup = 8 waves = 128 queries of one head. Per 32-key step the workgroup
// stages K(32x64) and V(64x32) tiles into double-buffered LDS via async loads;
// each wave runs 4 score WMMAs + online softmax + 4 PV WMMAs on the shared tiles.
struct AttnSmem {
  unsigned short K[2][32 * 64];   // [buf][key][d]
  unsigned short V[2][64 * 32];   // [buf][d][key]
  unsigned short P[8][16 * 32];   // per-wave P tile bounce
};

__device__ __forceinline__ void stage_tiles(const unsigned short* __restrict__ QKt,
                                            const unsigned short* __restrict__ Vbuf,
                                            AttnSmem* sm, int buf, int kb, int h,
                                            int wave, int lane) {
  {
    const int row = wave * 4 + (lane >> 3);   // 32 key rows, 128B each
    const int chunk = lane & 7;
    const unsigned short* g = QKt + (size_t)(kb + row) * 1024 + 512 + h * 64 + chunk * 8;
    unsigned loff = (unsigned)(uintptr_t)&sm->K[buf][row * 64 + chunk * 8];
    async_load_b128(loff, g);
  }
  {
    const int row = wave * 8 + (lane >> 2);   // 64 d rows, 64B each
    const int chunk = lane & 3;
    const unsigned short* g = Vbuf + (size_t)(h * 64 + row) * N_TOK + kb + chunk * 8;
    unsigned loff = (unsigned)(uintptr_t)&sm->V[buf][row * 32 + chunk * 8];
    async_load_b128(loff, g);
  }
}

__global__ void __launch_bounds__(256) k_attention(
    const unsigned short* __restrict__ QKt,  // 4096x1024 (cols 0..511 = Q, 512..1023 = K)
    const unsigned short* __restrict__ Vbuf, // 512x4096
    unsigned short* __restrict__ OutT) {     // 4096x512
  __shared__ __align__(16) AttnSmem sm;
  const int lane = threadIdx.x & 31;
  const int wave = threadIdx.x >> 5;
  const int h = blockIdx.x >> 5;                       // 32 blocks per head
  const int q0 = ((blockIdx.x & 31) * 8 + wave) * 16;  // 16 queries per wave
  const int half = lane >> 4;
  const float scale = 0.125f;                          // 1/sqrt(64)

  v16bf a0 = load_frag_A(QKt, 1024, q0, h * 64, lane);
  v16bf a1 = load_frag_A(QKt, 1024, q0, h * 64 + 32, lane);

  float mrow[8], lsum[8];
  v8f acc[4];
  v8f vz = {};
  #pragma unroll
  for (int r = 0; r < 8; ++r) { mrow[r] = -1e30f; lsum[r] = 0.f; }
  #pragma unroll
  for (int t = 0; t < 4; ++t) acc[t] = vz;

  unsigned short* pb = &sm.P[wave][0];

  // prologue: stage tile 0
  stage_tiles(QKt, Vbuf, &sm, 0, 0, h, wave, lane);
  wait_async0();
  __syncthreads();

  #pragma unroll 1
  for (int it = 0; it < N_TOK / 32; ++it) {
    const int cur = it & 1;
    if (it + 1 < N_TOK / 32)   // prefetch next 32-key tile into other buffer
      stage_tiles(QKt, Vbuf, &sm, cur ^ 1, (it + 1) * 32, h, wave, lane);

    const unsigned short* Kb = &sm.K[cur][0];
    const unsigned short* Vb = &sm.V[cur][0];

    v8f s0 = vz, s1 = vz;
    {
      v16bf b00 = load_frag_T(Kb, 64, 0,  0,  lane);
      v16bf b01 = load_frag_T(Kb, 64, 0,  32, lane);
      v16bf b10 = load_frag_T(Kb, 64, 16, 0,  lane);
      v16bf b11 = load_frag_T(Kb, 64, 16, 32, lane);
      s0 = wmma_bf16(a0, b00, s0);
      s0 = wmma_bf16(a1, b01, s0);
      s1 = wmma_bf16(a0, b10, s1);
      s1 = wmma_bf16(a1, b11, s1);
    }
    float p0[8], p1[8], alpha[8];
    #pragma unroll
    for (int r = 0; r < 8; ++r) {
      float v0 = s0[r] * scale;
      float v1 = s1[r] * scale;
      float mv = fmaxf(v0, v1);
      #pragma unroll
      for (int off = 8; off >= 1; off >>= 1)
        mv = fmaxf(mv, __shfl_xor(mv, off, 16));     // row max within 16-lane half
      float mnew = fmaxf(mrow[r], mv);
      alpha[r] = __expf(mrow[r] - mnew);
      p0[r] = __expf(v0 - mnew);
      p1[r] = __expf(v1 - mnew);
      lsum[r] = lsum[r] * alpha[r] + p0[r] + p1[r];  // per-lane partial row sum
      mrow[r] = mnew;
    }
    #pragma unroll
    for (int t = 0; t < 4; ++t)
      #pragma unroll
      for (int r = 0; r < 8; ++r) acc[t][r] *= alpha[r];
    // C-layout P -> per-wave LDS slice, reload in A-layout
    {
      const int key = lane & 15;
      #pragma unroll
      for (int r = 0; r < 8; ++r) {
        const int M = half * 8 + r;
        pb[M * 32 + key]      = f32_to_bf16(p0[r]);
        pb[M * 32 + 16 + key] = f32_to_bf16(p1[r]);
      }
    }
    __builtin_amdgcn_wave_barrier();
    asm volatile("s_wait_dscnt 0" ::: "memory");
    v16bf pa;
    {
      const int m = lane & 15;
      const unsigned short* p = pb + m * 32 + half * 8;
      FragU f;
      f.q[0] = *(const uint4*)(p);
      f.q[1] = *(const uint4*)(p + 16);
      pa = f.v;
    }
    #pragma unroll
    for (int t = 0; t < 4; ++t) {
      v16bf bv = load_frag_T(Vb, 32, t * 16, 0, lane);
      acc[t] = wmma_bf16(pa, bv, acc[t]);
    }
    // my async prefetch done + everyone done reading current buffers
    wait_async0();
    __syncthreads();
  }

  // finalize: reduce row sums across the 16-lane half, normalize, store bf16
  #pragma unroll
  for (int r = 0; r < 8; ++r) {
    float l = lsum[r];
    #pragma unroll
    for (int off = 8; off >= 1; off >>= 1)
      l += __shfl_xor(l, off, 16);
    lsum[r] = 1.0f / l;
  }
  const int ncol = lane & 15;
  #pragma unroll
  for (int t = 0; t < 4; ++t)
    #pragma unroll
    for (int r = 0; r < 8; ++r) {
      const int M = half * 8 + r;
      OutT[(size_t)(q0 + M) * CMID + h * 64 + t * 16 + ncol] =
          f32_to_bf16(acc[t][r] * lsum[r]);
    }
}

// ---------------- output projection (32x32 per wave) ----------------
// out[o][n] = sum_c Wo[o][c] * OutT[n][c], fp32 result in reference layout.
__global__ void __launch_bounds__(128) k_gemm_out(
    const unsigned short* __restrict__ Wo,    // 256x512
    const unsigned short* __restrict__ OutT,  // 4096x512
    float* __restrict__ out) {                // 256x4096
  const int lane = threadIdx.x & 31;
  const int wave = threadIdx.x >> 5;
  const int tile = blockIdx.x * 4 + wave;     // 8 x 128 tiles of 32x32
  const int oBase = (tile >> 7) * 32;
  const int n0 = (tile & 127) * 32;
  v8f acc[2][2];
  v8f vz = {};
  acc[0][0] = vz; acc[0][1] = vz; acc[1][0] = vz; acc[1][1] = vz;
  #pragma unroll
  for (int kb = 0; kb < CMID; kb += 32) {
    v16bf a0 = load_frag_A(Wo, CMID, oBase,      kb, lane);
    v16bf a1 = load_frag_A(Wo, CMID, oBase + 16, kb, lane);
    v16bf b0 = load_frag_T(OutT, CMID, n0,      kb, lane);
    v16bf b1 = load_frag_T(OutT, CMID, n0 + 16, kb, lane);
    acc[0][0] = wmma_bf16(a0, b0, acc[0][0]);
    acc[0][1] = wmma_bf16(a0, b1, acc[0][1]);
    acc[1][0] = wmma_bf16(a1, b0, acc[1][0]);
    acc[1][1] = wmma_bf16(a1, b1, acc[1][1]);
  }
  const int half = lane >> 4;
  #pragma unroll
  for (int i = 0; i < 2; ++i)
    #pragma unroll
    for (int j = 0; j < 2; ++j) {
      const int n = n0 + j * 16 + (lane & 15);
      #pragma unroll
      for (int r = 0; r < 8; ++r)
        out[(size_t)(oBase + i * 16 + half * 8 + r) * N_TOK + n] = acc[i][j][r];
    }
}

// ---------------- host launch ----------------

extern "C" void kernel_launch(void* const* d_in, const int* in_sizes, int n_in,
                              void* d_out, int out_size, void* d_ws, size_t ws_size,
                              hipStream_t stream) {
  (void)in_sizes; (void)n_in; (void)out_size; (void)ws_size;
  const float* x    = (const float*)d_in[0];  // 256*4096
  const float* Wqkv = (const float*)d_in[1];  // 1536*256
  const float* Wout = (const float*)d_in[2];  // 256*512

  char* ws = (char*)d_ws;
  unsigned short* Xt   = (unsigned short*)(ws + 0);          // 4096x256  (2 MB)
  unsigned short* Wqb  = (unsigned short*)(ws + 2097152);    // 1536x256  (768 KB)
  unsigned short* Wob  = (unsigned short*)(ws + 2883584);    // 256x512   (256 KB)
  unsigned short* QKt  = (unsigned short*)(ws + 3145728);    // 4096x1024 (8 MB)
  unsigned short* Vbuf = (unsigned short*)(ws + 11534336);   // 512x4096  (4 MB)
  unsigned short* OutT = (unsigned short*)(ws + 15728640);   // 4096x512  (4 MB)

  k_transpose_x_bf16<<<4096, 256, 0, stream>>>(x, Xt);
  k_cvt_bf16<<<1536, 256, 0, stream>>>(Wqkv, Wqb, OQKV * CIN);
  k_cvt_bf16<<<512, 256, 0, stream>>>(Wout, Wob, 256 * CMID);
  k_gemm_qkv<<<1536, 128, 0, stream>>>(Wqb, Xt, QKt, Vbuf);
  k_attention<<<256, 256, 0, stream>>>(QKt, Vbuf, OutT);
  k_gemm_out<<<256, 128, 0, stream>>>(Wob, OutT, (float*)d_out);
}
